// ShapeNetModel_25314537242845
// MI455X (gfx1250) — matrix operand order repaired and checked by reading, not compiled, not run on hardware
//
#include <hip/hip_runtime.h>

// ---------------------------------------------------------------------------
// ShapeNet part-seg network on MI455X (gfx1250, wave32, WMMA bf16).
// All heavy math runs through V_WMMA_F32_16X16X32_BF16 (fp32 accumulate).
// Pure-copy LDS staging uses GLOBAL_LOAD_ASYNC_TO_LDS_B128 (ASYNCcnt path).
//
// Input order (setup_inputs dict order, nested dicts flattened in insertion
// order):
//   0 x[8,2048,3] f32, 1 category_id[8,16] f32,
//   2 emb1_w[64,3], 3 emb1_g, 4 emb1_b, 5 emb2_w[128,64], 6 emb2_g, 7 emb2_b,
//   8..13  ca1 {q,k,v,ff,g,b}, 14..19 ca2,
//   20..25 p2p1, 26..31 p2p2, 32..37 p2p3, 38..43 p2p4,
//   44 fuse_w[1024,512], 45 fuse_g, 46 fuse_b,
//   47 cat_w[64,16], 48 cat_g, 49 cat_b,
//   50 c1_w[512,3136], 51 c1_g, 52 c1_b,
//   53 c2_w[256,512], 54 c2_g, 55 c2_b, 56 c3_w[50,256]
// ---------------------------------------------------------------------------

typedef unsigned short u16;
typedef unsigned int   u32;
typedef unsigned long long u64;
typedef __attribute__((ext_vector_type(16))) __bf16 v16bf;
typedef __attribute__((ext_vector_type(8)))  float  v8f;

static constexpr int BB   = 8;
static constexpr int NN   = 2048;
static constexpr int MTOT = BB * NN;   // 16384 columns in every activation

union FragBF { v16bf v; u32 u[8]; u16 s[16]; };

__device__ __forceinline__ u16 f2b(float f) {
    u32 u = __float_as_uint(f);
    u32 r = (u + 0x7FFFu + ((u >> 16) & 1u)) >> 16;   // round-to-nearest-even
    return (u16)r;
}
__device__ __forceinline__ float b2f(u16 h) {
    return __uint_as_float(((u32)h) << 16);
}

// Async cache->LDS 16B copy (GV mode, no VGPR round trip; tracked by ASYNCcnt).
// Generic LDS addr low 32 bits == wave-relative LDS byte offset.
__device__ __forceinline__ void async_b128(const void* lds_dst, const void* gsrc) {
    u32 lo = (u32)(size_t)lds_dst;
    u64 ga = (u64)(size_t)gsrc;
    asm volatile("global_load_async_to_lds_b128 %0, %1, off"
                 :: "v"(lo), "v"(ga) : "memory");
}
__device__ __forceinline__ void async_wait0() {
    asm volatile("s_wait_asynccnt 0x0" ::: "memory");
}

// ---------------------------------------------------------------------------
// Weight f32 -> bf16 with row restride + K zero-padding.
// ---------------------------------------------------------------------------
__global__ void wcvt_kernel(const float* __restrict__ src, u16* __restrict__ dst,
                            int O, int Ksrc, int Kcopy, int Kpad) {
    int i = blockIdx.x * 256 + threadIdx.x;
    if (i >= O * Kpad) return;
    int o = i / Kpad, k = i - o * Kpad;
    float v = (k < Kcopy) ? src[(size_t)o * Ksrc + k] : 0.f;
    dst[i] = f2b(v);
}

// xyz [B,N,3] f32 -> [32, M] bf16 (rows 3..31 zero) so K=32 for emb1 GEMM.
__global__ void xpose_kernel(const float* __restrict__ x, u16* __restrict__ xt) {
    int i = blockIdx.x * 256 + threadIdx.x;
    if (i >= 32 * MTOT) return;
    int r = i / MTOT, p = i - r * MTOT;
    float v = (r < 3) ? x[(size_t)p * 3 + r] : 0.f;
    xt[i] = f2b(v);
}

// ---------------------------------------------------------------------------
// Exact 32-NN per point via insertion list in LDS (VALU; <1% of FLOPs).
// Indices stored as *global* columns (b*N + m), ready for gathers.
// ---------------------------------------------------------------------------
__global__ __launch_bounds__(128) void knn_kernel(const float* __restrict__ xyz,
                                                  int* __restrict__ nbidx) {
    __shared__ float sd[128 * 32];
    __shared__ int   si[128 * 32];
    int t = threadIdx.x;
    int p = blockIdx.x * 128 + t;
    int b = p >> 11, base = b << 11;
    float x0 = xyz[(size_t)p * 3], x1 = xyz[(size_t)p * 3 + 1], x2 = xyz[(size_t)p * 3 + 2];
    float* d  = sd + t * 32;
    int*   id = si + t * 32;
    for (int j = 0; j < 32; ++j) { d[j] = 3.4e38f; id[j] = base; }
    for (int m = 0; m < NN; ++m) {
        const float* q = xyz + (size_t)(base + m) * 3;
        float dx = x0 - q[0], dy = x1 - q[1], dz = x2 - q[2];
        float dist = dx * dx + dy * dy + dz * dz;
        if (dist < d[31]) {
            int j = 31;
            while (j > 0 && d[j - 1] > dist) { d[j] = d[j - 1]; id[j] = id[j - 1]; --j; }
            d[j] = dist; id[j] = base + m;
        }
    }
    for (int j = 0; j < 32; ++j) nbidx[(size_t)p * 32 + j] = id[j];
}

// ---------------------------------------------------------------------------
// Tiled bf16 WMMA GEMM:  Y[O,M] = epilogue( W[O,K] * X[K,M] ).
// Block = 256 threads (8 waves), tile 64(O) x 128(cols); each wave owns
// 16x64 via 4 accumulators. W tile arrives via async-to-LDS (straight copy);
// the X tile is K-transposed through VGPRs so every WMMA fragment load is a
// contiguous-dword ds_load.
//   A frag (16x32 bf16): lane row = l15; elem e -> k = hi*8+e (e<8), 16+hi*8+e-8
//   B frag (32x16 bf16): lane col = l15; elem e -> k = hi*8+e (e<8), 16+hi*8+e-8
//   C/D  (16x16 f32):    lane col = l15; vgpr r -> row = r + 8*hi
// ---------------------------------------------------------------------------
enum { E_NONE = 0, E_BN_RELU = 1, E_BN_RELU_RES = 2, E_BN_LRELU = 3,
       E_BN_RELU_BIAS = 4, E_OUT_T = 5 };

template <int EPI>
__global__ __launch_bounds__(256) void gemm_bf16(
    const u16* __restrict__ Wb, const u16* __restrict__ Xb,
    u16* __restrict__ Yb, float* __restrict__ Yf,
    int O, int K,
    const float* __restrict__ gm, const float* __restrict__ bt,
    const u16* __restrict__ res, const float* __restrict__ biasBO) {
    __shared__ __align__(16) u16 Wl[64 * 32];    // [o][k]
    __shared__ __align__(16) u16 Xl[128 * 32];   // [col][k]  (k contiguous)
    int tid = threadIdx.x;
    int lane = tid & 31, wave = tid >> 5;
    int l15 = lane & 15, hi = lane >> 4;
    int wo = wave & 3, wc = wave >> 2;
    int colbase = blockIdx.x * 128;
    int obase = blockIdx.y * 64;
    v8f acc[4] = {};

    for (int k0 = 0; k0 < K; k0 += 32) {
        // ---- stage W tile [64 x 32]: async cache->LDS copy (row-guarded) ----
        {
            int r = tid >> 2, cq = (tid & 3) << 3;
            int o = obase + r;
            u16* dstp = Wl + r * 32 + cq;
            if (o < O) async_b128(dstp, Wb + (size_t)o * K + k0 + cq);
            else       *(uint4*)dstp = make_uint4(0u, 0u, 0u, 0u);
        }
        // ---- stage X tile transposed to [col][k] (manual; async can't transpose) ----
        for (int it = 0; it < 2; ++it) {
            int id = tid + it * 256;
            int kr = id >> 4;              // 0..31
            int c8 = (id & 15) << 3;       // 0..120
            uint4 xv = *(const uint4*)(Xb + (size_t)(k0 + kr) * MTOT + colbase + c8);
            __align__(16) u16 tmp[8];
            *(uint4*)tmp = xv;
#pragma unroll
            for (int i = 0; i < 8; ++i) Xl[(c8 + i) * 32 + kr] = tmp[i];
        }
        if (k0 + 32 < K && lane < 8)
            __builtin_prefetch(Xb + (size_t)(k0 + 32 + lane * 4) * MTOT + colbase, 0, 0);
        async_wait0();
        __syncthreads();

        FragBF a;
        {
            const u16* wr = Wl + (wo * 16 + l15) * 32;
#pragma unroll
            for (int j = 0; j < 4; ++j) {
                a.u[j]     = *(const u32*)(wr + hi * 8 + 2 * j);
                a.u[4 + j] = *(const u32*)(wr + 16 + hi * 8 + 2 * j);
            }
        }
#pragma unroll
        for (int s = 0; s < 4; ++s) {
            FragBF b;
            const u16* xr2 = Xl + (wc * 64 + s * 16 + l15) * 32;
#pragma unroll
            for (int j = 0; j < 8; ++j) b.u[j] = *(const u32*)(xr2 + hi * 8 + (j >> 2) * 16 + 2 * (j & 3));
            acc[s] = __builtin_amdgcn_wmma_f32_16x16x32_bf16(
                false, a.v, false, b.v, (short)0, acc[s], false, false);
        }
        __syncthreads();
    }

#pragma unroll
    for (int s = 0; s < 4; ++s) {
        int col = colbase + wc * 64 + s * 16 + l15;
#pragma unroll
        for (int r = 0; r < 8; ++r) {
            int o = obase + wo * 16 + r + hi * 8;
            if (o >= O) continue;
            float v = acc[s][r];
            if (EPI == E_BN_RELU) {
                v = fmaxf(gm[o] * v + bt[o], 0.f);
            } else if (EPI == E_BN_RELU_RES) {
                v = fmaxf(gm[o] * v + bt[o], 0.f) + b2f(res[(size_t)o * MTOT + col]);
            } else if (EPI == E_BN_LRELU) {
                float t2 = gm[o] * v + bt[o];
                v = (t2 > 0.f) ? t2 : 0.2f * t2;
            } else if (EPI == E_BN_RELU_BIAS) {
                int bidx = col >> 11;   // batch
                v = fmaxf(gm[o] * (v + biasBO[bidx * O + o]) + bt[o], 0.f);
            }
            if (EPI == E_OUT_T) Yf[(size_t)col * 50 + o] = v;   // [B,N,50]
            else                Yb[(size_t)o * MTOT + col] = f2b(v);
        }
    }
}

// ---------------------------------------------------------------------------
// Neighbor cross-attention core. conv(gather(h)) == gather(conv(h)), so
// Q/K/V are full GEMM outputs; here: per point (one wave) attend over its
// 32 neighbors (one per lane), then write diff = h - attn_out for the ff GEMM.
// ---------------------------------------------------------------------------
__global__ __launch_bounds__(256) void neigh_attn(
    const u16* __restrict__ Q, const u16* __restrict__ Kx, const u16* __restrict__ V,
    const int* __restrict__ nb, const u16* __restrict__ H, u16* __restrict__ D) {
    __shared__ float attL[8][32];
    __shared__ int   nbL[8][32];
    int w = threadIdx.x >> 5, lane = threadIdx.x & 31;
    int p = blockIdx.x * 8 + w;
    int nj = nb[(size_t)p * 32 + lane];
    float s = 0.f;
    for (int c = 0; c < 128; ++c)
        s += b2f(Q[(size_t)c * MTOT + p]) * b2f(Kx[(size_t)c * MTOT + nj]);
    s *= 0.08838834764831845f;   // 1/sqrt(128)
    float mx = s;
    for (int d2 = 16; d2 >= 1; d2 >>= 1) mx = fmaxf(mx, __shfl_xor(mx, d2, 32));
    float e = __expf(s - mx), sum = e;
    for (int d2 = 16; d2 >= 1; d2 >>= 1) sum += __shfl_xor(sum, d2, 32);
    attL[w][lane] = e / sum;
    nbL[w][lane] = nj;
    __syncthreads();
    int c0 = lane * 4;
    float a0 = 0.f, a1 = 0.f, a2 = 0.f, a3 = 0.f;
    for (int j = 0; j < 32; ++j) {
        float a = attL[w][j];
        int n2 = nbL[w][j];
        a0 += a * b2f(V[(size_t)(c0 + 0) * MTOT + n2]);
        a1 += a * b2f(V[(size_t)(c0 + 1) * MTOT + n2]);
        a2 += a * b2f(V[(size_t)(c0 + 2) * MTOT + n2]);
        a3 += a * b2f(V[(size_t)(c0 + 3) * MTOT + n2]);
    }
    D[(size_t)(c0 + 0) * MTOT + p] = f2b(b2f(H[(size_t)(c0 + 0) * MTOT + p]) - a0);
    D[(size_t)(c0 + 1) * MTOT + p] = f2b(b2f(H[(size_t)(c0 + 1) * MTOT + p]) - a1);
    D[(size_t)(c0 + 2) * MTOT + p] = f2b(b2f(H[(size_t)(c0 + 2) * MTOT + p]) - a2);
    D[(size_t)(c0 + 3) * MTOT + p] = f2b(b2f(H[(size_t)(c0 + 3) * MTOT + p]) - a3);
}

// ---------------------------------------------------------------------------
// Flash-style full p2p attention: block = 4 waves, each wave owns 16 queries;
// 32-key chunks staged in LDS cooperatively (V via async-to-LDS, K manually
// transposed); S = QtK and out += V*Pt both WMMA. Online softmax in fp32;
// no NxN matrix ever touches HBM.
// ---------------------------------------------------------------------------
__global__ __launch_bounds__(128) void p2p_flash(
    const u16* __restrict__ Q, const u16* __restrict__ Kx,
    const u16* __restrict__ V, u16* __restrict__ XR) {
    __shared__ __align__(16) u16 Qs[4][16][128];  // [wave][q][c]
    __shared__ __align__(16) u16 Ks[32][128];     // [key][c]
    __shared__ __align__(16) u16 Vs[128][32];     // [c][key]
    __shared__ __align__(16) u16 Ps[4][16][32];   // [wave][q][key]
    __shared__ float stats[4][16];
    const float SC = 0.08838834764831845f;
    int tid = threadIdx.x, w = tid >> 5, lane = tid & 31;
    int l15 = lane & 15, hi = lane >> 4;
    int bb = blockIdx.y;
    int qbase = blockIdx.x * 64 + w * 16;
    int col0 = bb * NN + qbase;

    for (int cc = 0; cc < 64; ++cc) {             // coalesced Q tile load
        int c = cc * 2 + hi;
        Qs[w][l15][c] = Q[(size_t)c * MTOT + col0 + l15];
    }
    v8f out[8] = {};
    float mrow[8], lrow[8];
#pragma unroll
    for (int r = 0; r < 8; ++r) { mrow[r] = -1e30f; lrow[r] = 0.f; }

    for (int kc = 0; kc < NN / 32; ++kc) {
        int keyb = bb * NN + kc * 32;
        __syncthreads();
        {   // stage K (transposed) and V (async straight copy), one channel/thread
            int c = tid;
            const u16* kp = Kx + (size_t)c * MTOT + keyb;
            const u16* vp = V + (size_t)c * MTOT + keyb;
            async_b128(&Vs[c][0],  vp);
            async_b128(&Vs[c][8],  vp + 8);
            async_b128(&Vs[c][16], vp + 16);
            async_b128(&Vs[c][24], vp + 24);
            __align__(16) u16 tmp[32];
            *(uint4*)&tmp[0]  = *(const uint4*)(kp);
            *(uint4*)&tmp[8]  = *(const uint4*)(kp + 8);
            *(uint4*)&tmp[16] = *(const uint4*)(kp + 16);
            *(uint4*)&tmp[24] = *(const uint4*)(kp + 24);
#pragma unroll
            for (int j = 0; j < 32; ++j) Ks[j][c] = tmp[j];
        }
        async_wait0();
        __syncthreads();

        v8f S0 = {}, S1 = {};
#pragma unroll
        for (int ct = 0; ct < 4; ++ct) {          // K-dim = channels, 4x32
            int cb = ct * 32;
            FragBF a, b0, b1;
            const u16* qrow = &Qs[w][l15][0];
#pragma unroll
            for (int j = 0; j < 4; ++j) {
                a.u[j]     = *(const u32*)(qrow + cb + hi * 8 + 2 * j);
                a.u[4 + j] = *(const u32*)(qrow + cb + 16 + hi * 8 + 2 * j);
            }
#pragma unroll
            for (int j = 0; j < 8; ++j) {
                b0.u[j] = *(const u32*)(&Ks[l15][0] + cb + hi * 8 + (j >> 2) * 16 + 2 * (j & 3));
                b1.u[j] = *(const u32*)(&Ks[l15 + 16][0] + cb + hi * 8 + (j >> 2) * 16 + 2 * (j & 3));
            }
            S0 = __builtin_amdgcn_wmma_f32_16x16x32_bf16(false, a.v, false, b0.v, (short)0, S0, false, false);
            S1 = __builtin_amdgcn_wmma_f32_16x16x32_bf16(false, a.v, false, b1.v, (short)0, S1, false, false);
        }
        // online softmax; row q = r + 8*hi lives across the 16 lanes of a half
        float facl[8];
#pragma unroll
        for (int r = 0; r < 8; ++r) {
            float s0 = S0[r] * SC, s1 = S1[r] * SC;
            float mx = fmaxf(s0, s1);
            for (int d2 = 8; d2 >= 1; d2 >>= 1) mx = fmaxf(mx, __shfl_xor(mx, d2, 32));
            float nm = fmaxf(mrow[r], mx);
            float p0 = __expf(s0 - nm), p1 = __expf(s1 - nm);
            float rs = p0 + p1;
            for (int d2 = 8; d2 >= 1; d2 >>= 1) rs += __shfl_xor(rs, d2, 32);
            float fac = __expf(mrow[r] - nm);
            lrow[r] = lrow[r] * fac + rs;
            mrow[r] = nm;
            Ps[w][r + 8 * hi][l15]      = f2b(p0);
            Ps[w][r + 8 * hi][l15 + 16] = f2b(p1);
            facl[r] = fac;
        }
        if (l15 == 0) {
#pragma unroll
            for (int r = 0; r < 8; ++r) stats[w][r + 8 * hi] = facl[r];
        }
        asm volatile("" ::: "memory");            // same-wave LDS ops are in-order
        float facq = stats[w][l15];               // factor for this lane's q column
#pragma unroll
        for (int i = 0; i < 8; ++i) out[i] *= facq;
#pragma unroll
        for (int ct = 0; ct < 8; ++ct) {          // out[c,q] += V * P^T, k = 32 keys
            FragBF a, b;
            const u16* vrow = &Vs[ct * 16 + l15][0];
#pragma unroll
            for (int j = 0; j < 4; ++j) {
                a.u[j]     = *(const u32*)(vrow + hi * 8 + 2 * j);
                a.u[4 + j] = *(const u32*)(vrow + 16 + hi * 8 + 2 * j);
            }
            const u16* prow = &Ps[w][l15][0];
#pragma unroll
            for (int j = 0; j < 8; ++j) b.u[j] = *(const u32*)(prow + hi * 8 + (j >> 2) * 16 + 2 * (j & 3));
            out[ct] = __builtin_amdgcn_wmma_f32_16x16x32_bf16(false, a.v, false, b.v, (short)0, out[ct], false, false);
        }
    }
    if (l15 == 0) {
#pragma unroll
        for (int r = 0; r < 8; ++r) stats[w][r + 8 * hi] = lrow[r];
    }
    asm volatile("" ::: "memory");
    float linv = 1.0f / stats[w][l15];
#pragma unroll
    for (int ct = 0; ct < 8; ++ct)
#pragma unroll
        for (int r = 0; r < 8; ++r) {
            int c = ct * 16 + r + 8 * hi;
            XR[(size_t)c * MTOT + col0 + l15] = f2b(out[ct][r] * linv);
        }
}

__global__ void diff_kernel(const u16* __restrict__ H, const u16* __restrict__ XR,
                            u16* __restrict__ D, int n) {
    int i = blockIdx.x * 256 + threadIdx.x;
    if (i < n) D[i] = f2b(b2f(H[i]) - b2f(XR[i]));
}

// max / mean over N per (o, batch)
__global__ __launch_bounds__(256) void pool_kernel(const u16* __restrict__ F,
                                                   float* __restrict__ fmax,
                                                   float* __restrict__ favg) {
    int o = blockIdx.x, b = blockIdx.y, t = threadIdx.x;
    const u16* row = F + (size_t)o * MTOT + b * NN;
    float mx = -3.4e38f, sm = 0.f;
    for (int i = t; i < NN; i += 256) {
        float v = b2f(row[i]);
        mx = fmaxf(mx, v); sm += v;
    }
    for (int d2 = 16; d2 >= 1; d2 >>= 1) {
        mx = fmaxf(mx, __shfl_xor(mx, d2, 32));
        sm += __shfl_xor(sm, d2, 32);
    }
    __shared__ float smx[8], ssm[8];
    if ((t & 31) == 0) { smx[t >> 5] = mx; ssm[t >> 5] = sm; }
    __syncthreads();
    if (t == 0) {
        float M2 = smx[0], S2 = ssm[0];
        for (int i = 1; i < 8; ++i) { M2 = fmaxf(M2, smx[i]); S2 += ssm[i]; }
        fmax[b * 1024 + o] = M2;
        favg[b * 1024 + o] = S2 * (1.f / NN);
    }
}

// cat = lrelu(bn(Wcat * onehot)) -> [8,64] f32
__global__ void cat_kernel(const float* __restrict__ cid, const float* __restrict__ Wc,
                           const float* __restrict__ g, const float* __restrict__ be,
                           float* __restrict__ out) {
    int i = blockIdx.x * 64 + threadIdx.x;
    if (i >= 8 * 64) return;
    int b = i >> 6, o = i & 63;
    float s = 0.f;
    for (int j = 0; j < 16; ++j) s += Wc[o * 16 + j] * cid[b * 16 + j];
    float t2 = g[o] * s + be[o];
    out[i] = (t2 > 0.f) ? t2 : 0.2f * t2;
}

// Fold the broadcast part of c1 (fmax/favg/cat channels) into per-batch bias:
// bias1[b,o] = W1[o,1024:2048].fmax + W1[o,2048:3072].favg + W1[o,3072:3136].cat
__global__ __launch_bounds__(256) void bias1_kernel(
    const float* __restrict__ W1, const float* __restrict__ fmax,
    const float* __restrict__ favg, const float* __restrict__ cat64,
    float* __restrict__ bias1) {
    int i = blockIdx.x * 256 + threadIdx.x;
    if (i >= 8 * 512) return;
    int b = i >> 9, o = i & 511;
    const float* wr = W1 + (size_t)o * 3136;
    float s = 0.f;
    for (int j = 0; j < 1024; ++j) s += wr[1024 + j] * fmax[b * 1024 + j];
    for (int j = 0; j < 1024; ++j) s += wr[2048 + j] * favg[b * 1024 + j];
    for (int j = 0; j < 64; ++j)   s += wr[3072 + j] * cat64[b * 64 + j];
    bias1[i] = s;
}

// ---------------------------------------------------------------------------
extern "C" void kernel_launch(void* const* d_in, const int* in_sizes, int n_in,
                              void* d_out, int out_size, void* d_ws, size_t ws_size,
                              hipStream_t stream) {
    (void)in_sizes; (void)n_in; (void)out_size; (void)ws_size;
    const float* x   = (const float*)d_in[0];
    const float* cid = (const float*)d_in[1];
    auto F = [&](int i) { return (const float*)d_in[i]; };

    size_t off = 0;
    auto alloc = [&](size_t bytes) -> void* {
        void* p = (char*)d_ws + off;
        off += (bytes + 255) & ~(size_t)255;
        return p;
    };
    int* nbidx   = (int*)alloc((size_t)MTOT * 32 * 4);
    u16* xt      = (u16*)alloc((size_t)32 * MTOT * 2);
    u16* w_emb1  = (u16*)alloc((size_t)64 * 32 * 2);
    u16* w_emb2  = (u16*)alloc((size_t)128 * 64 * 2);
    u16* w_att[24];
    for (int i = 0; i < 24; ++i) w_att[i] = (u16*)alloc((size_t)128 * 128 * 2);
    u16* w_fuse  = (u16*)alloc((size_t)1024 * 512 * 2);
    u16* w_c1    = (u16*)alloc((size_t)512 * 1024 * 2);
    u16* w_c2    = (u16*)alloc((size_t)256 * 512 * 2);
    u16* w_c3    = (u16*)alloc((size_t)50 * 256 * 2);
    u16* h64     = (u16*)alloc((size_t)64 * MTOT * 2);
    u16* hA      = (u16*)alloc((size_t)128 * MTOT * 2);
    u16* hB      = (u16*)alloc((size_t)128 * MTOT * 2);
    u16* Qb      = (u16*)alloc((size_t)128 * MTOT * 2);
    u16* Kb      = (u16*)alloc((size_t)128 * MTOT * 2);
    u16* Vb      = (u16*)alloc((size_t)128 * MTOT * 2);
    u16* Db      = (u16*)alloc((size_t)128 * MTOT * 2);
    u16* XRb     = (u16*)alloc((size_t)128 * MTOT * 2);
    u16* fbuf    = (u16*)alloc((size_t)512 * MTOT * 2);
    u16* fused   = (u16*)alloc((size_t)1024 * MTOT * 2);
    u16* c1o     = (u16*)alloc((size_t)512 * MTOT * 2);
    u16* c2o     = (u16*)alloc((size_t)256 * MTOT * 2);
    float* fmaxb = (float*)alloc(8 * 1024 * 4);
    float* favgb = (float*)alloc(8 * 1024 * 4);
    float* cat64 = (float*)alloc(8 * 64 * 4);
    float* bias1 = (float*)alloc(8 * 512 * 4);

    auto cvt = [&](const float* s, u16* dst, int O, int Ks, int Kc, int Kp) {
        int tot = O * Kp;
        wcvt_kernel<<<(tot + 255) / 256, 256, 0, stream>>>(s, dst, O, Ks, Kc, Kp);
    };
    cvt(F(2), w_emb1, 64, 3, 3, 32);
    cvt(F(5), w_emb2, 128, 64, 64, 64);
    for (int j = 0; j < 6; ++j) {
        int base = (j < 2) ? (8 + 6 * j) : (20 + 6 * (j - 2));
        for (int m = 0; m < 4; ++m) cvt(F(base + m), w_att[j * 4 + m], 128, 128, 128, 128);
    }
    cvt(F(44), w_fuse, 1024, 512, 512, 512);
    cvt(F(50), w_c1, 512, 3136, 1024, 1024);
    cvt(F(53), w_c2, 256, 512, 512, 512);
    cvt(F(56), w_c3, 50, 256, 256, 256);
    xpose_kernel<<<(32 * MTOT + 255) / 256, 256, 0, stream>>>(x, xt);
    knn_kernel<<<MTOT / 128, 128, 0, stream>>>(x, nbidx);

    dim3 blk(256);
    auto G = [&](int O) { return dim3(MTOT / 128, (O + 63) / 64); };

    // embedding 3 -> 64 -> 128
    gemm_bf16<E_BN_RELU><<<G(64), blk, 0, stream>>>(w_emb1, xt, h64, nullptr, 64, 32,
                                                    F(3), F(4), nullptr, nullptr);
    gemm_bf16<E_BN_RELU><<<G(128), blk, 0, stream>>>(w_emb2, h64, hA, nullptr, 128, 64,
                                                     F(6), F(7), nullptr, nullptr);

    u16* cur = hA;
    u16* nxt = hB;
    // two neighbor cross-attention blocks
    for (int j = 0; j < 2; ++j) {
        int base = 8 + 6 * j;
        gemm_bf16<E_NONE><<<G(128), blk, 0, stream>>>(w_att[j * 4 + 0], cur, Qb, nullptr, 128, 128, nullptr, nullptr, nullptr, nullptr);
        gemm_bf16<E_NONE><<<G(128), blk, 0, stream>>>(w_att[j * 4 + 1], cur, Kb, nullptr, 128, 128, nullptr, nullptr, nullptr, nullptr);
        gemm_bf16<E_NONE><<<G(128), blk, 0, stream>>>(w_att[j * 4 + 2], cur, Vb, nullptr, 128, 128, nullptr, nullptr, nullptr, nullptr);
        neigh_attn<<<MTOT / 8, 256, 0, stream>>>(Qb, Kb, Vb, nbidx, cur, Db);
        gemm_bf16<E_BN_RELU_RES><<<G(128), blk, 0, stream>>>(w_att[j * 4 + 3], Db, nxt, nullptr, 128, 128, F(base + 4), F(base + 5), cur, nullptr);
        u16* t = cur; cur = nxt; nxt = t;
    }
    // four full p2p attention blocks; outputs land in the concat slab fbuf
    for (int j = 0; j < 4; ++j) {
        int base = 20 + 6 * j;
        int wb = (2 + j) * 4;
        gemm_bf16<E_NONE><<<G(128), blk, 0, stream>>>(w_att[wb + 0], cur, Qb, nullptr, 128, 128, nullptr, nullptr, nullptr, nullptr);
        gemm_bf16<E_NONE><<<G(128), blk, 0, stream>>>(w_att[wb + 1], cur, Kb, nullptr, 128, 128, nullptr, nullptr, nullptr, nullptr);
        gemm_bf16<E_NONE><<<G(128), blk, 0, stream>>>(w_att[wb + 2], cur, Vb, nullptr, 128, 128, nullptr, nullptr, nullptr, nullptr);
        p2p_flash<<<dim3(NN / 64, BB), 128, 0, stream>>>(Qb, Kb, Vb, XRb);
        diff_kernel<<<(128 * MTOT + 255) / 256, 256, 0, stream>>>(cur, XRb, Db, 128 * MTOT);
        u16* dst = fbuf + (size_t)j * 128 * MTOT;
        gemm_bf16<E_BN_RELU_RES><<<G(128), blk, 0, stream>>>(w_att[wb + 3], Db, dst, nullptr, 128, 128, F(base + 4), F(base + 5), cur, nullptr);
        cur = dst;
    }
    // fuse 512 -> 1024 with leaky-relu
    gemm_bf16<E_BN_LRELU><<<G(1024), blk, 0, stream>>>(w_fuse, fbuf, fused, nullptr, 1024, 512, F(45), F(46), nullptr, nullptr);
    pool_kernel<<<dim3(1024, 8), 256, 0, stream>>>(fused, fmaxb, favgb);
    cat_kernel<<<8, 64, 0, stream>>>(cid, F(47), F(48), F(49), cat64);
    bias1_kernel<<<(8 * 512 + 255) / 256, 256, 0, stream>>>(F(50), fmaxb, favgb, cat64, bias1);
    // c1 (K=1024 GEMM + folded per-batch bias), c2, c3 (transposed f32 out)
    gemm_bf16<E_BN_RELU_BIAS><<<G(512), blk, 0, stream>>>(w_c1, fused, c1o, nullptr, 512, 1024, F(51), F(52), nullptr, bias1);
    gemm_bf16<E_BN_RELU><<<G(256), blk, 0, stream>>>(w_c2, c1o, c2o, nullptr, 256, 512, F(53), F(54), nullptr, nullptr);
    gemm_bf16<E_OUT_T><<<G(50), blk, 0, stream>>>(w_c3, c2o, nullptr, (float*)d_out, 50, 256, nullptr, nullptr, nullptr, nullptr);
}
